// CrystalHypergraphConv_55886114455552
// MI455X (gfx1250) — compile-verified, matrix operand order checked by reading, not run on hardware
//
#include <hip/hip_runtime.h>
#include <hip/hip_bf16.h>
#include <math.h>

// ---------------------------------------------------------------------------
// CrystalHypergraphConv for MI455X (gfx1250, wave32, WMMA)
// GEMMs ([rows,128] @ [128,64]) on the matrix pipe via V_WMMA_F32_16X16X32_BF16.
// Weights staged in LDS pre-packed in B-fragment layout so each fragment is a
// single aligned 32B LDS read per lane (ds_load_b128 x2, no b16 repacking).
// ---------------------------------------------------------------------------

#define Nn      50000
#define Bb      1000
#define EBn     300000
#define Mm      50000
#define NLAYERS 6
#define EPSV    1e-5f

typedef __attribute__((ext_vector_type(16))) __bf16 v16bf;
typedef __attribute__((ext_vector_type(8)))  float  v8f;

__device__ __forceinline__ float sigmoidf_(float v) {
    return 1.0f / (1.0f + __expf(-v));
}
__device__ __forceinline__ float softplusf_(float v) {
    return v > 20.0f ? v : log1pf(__expf(v));
}

// ---------------------------------------------------------------------------
// Weight staging: convert f32 K-major [128,64] weights into LDS laid out as
// WMMA B fragments: sP[frag][lane][j], frag = kk*4+nt (K-tile, N-tile).
//  Lane L (0..15):  col N = nt*16+L,      K = {k0+0..7,  k0+16..23}
//  Lane L (16..31): col N = nt*16+(L-16), K = {k0+8..15, k0+24..31}
// Each lane's 16 bf16 values are contiguous (32 B) -> ds_load_b128 x2.
// ---------------------------------------------------------------------------
__device__ __forceinline__ void stage_weights_packed(const float* __restrict__ W,
                                                     __bf16* sP, int tid) {
    for (int idx = tid; idx < 16 * 32 * 16; idx += 256) {
        int f = idx >> 9;          // fragment id 0..15
        int l = (idx >> 4) & 31;   // lane
        int j = idx & 15;          // element within fragment
        int kk = f >> 2, nt = f & 3;
        int n = nt * 16 + (l & 15);
        int k = kk * 32 + ((l & 16) ? 8 : 0) + (j < 8 ? j : j + 8);
        sP[idx] = (__bf16)W[k * 64 + n];
    }
}

__device__ __forceinline__ v16bf load_b_packed(const __bf16* sP, int kk, int nt, int lane) {
    return *((const v16bf*)(sP + (((kk * 4 + nt) * 32 + lane) << 4)));
}

// A fragment (16x32, MxK) for row = concat(ra[0..63], rb[0..63]), f32->bf16.
// Per lane this reads two contiguous 32B spans -> global_load_b128 + cvt_pk.
__device__ __forceinline__ v16bf load_a_concat(const float* __restrict__ ra,
                                               const float* __restrict__ rb,
                                               int k0, int lane) {
    int kb = k0 + ((lane & 16) ? 8 : 0);
    v16bf a;
#pragma unroll
    for (int j = 0; j < 16; ++j) {
        int k = kb + (j < 8 ? j : j + 8);
        float v = (k < 64) ? ra[k] : rb[k - 64];
        a[j] = (__bf16)v;
    }
    return a;
}

// ---------------------------------------------------------------------------
// Generic small linear layer  Y[r,c] = act(sum_k X[r,k]*W[k,c] + b[c])
// act: 0 = identity, 1 = softplus. Memory-bound (embeddings + MLP head).
// ---------------------------------------------------------------------------
__global__ void lin_kernel(const float* __restrict__ X, const float* __restrict__ W,
                           const float* __restrict__ b, float* __restrict__ Y,
                           int rows, int Kin, int Kout, int act) {
    int i = blockIdx.x * blockDim.x + threadIdx.x;
    if (i >= rows * Kout) return;
    int r = i / Kout, c = i - r * Kout;
    const float* xr = X + (size_t)r * Kin;
    float s = b[c];
    for (int k = 0; k < Kin; ++k) s = fmaf(xr[k], W[k * Kout + c], s);
    if (act == 1) s = softplusf_(s);
    Y[i] = s;
}

// Count occurrences (launch-invariant segment counts).
__global__ void count_kernel(const int* __restrict__ idx, int n, float* __restrict__ cnt) {
    int i = blockIdx.x * blockDim.x + threadIdx.x;
    if (i >= n) return;
    atomicAdd(&cnt[idx[i]], 1.0f);
}

// Hedge aggregation: agg[e,c] = mean_d x[nodes[e*deg+d], c]
// (incidence is repeat(arange(E),deg) -> contiguous, fixed degree, no atomics)
__global__ void agg_kernel(const float* __restrict__ x, const int* __restrict__ nodes,
                           float* __restrict__ agg, int E, int deg) {
    int i = blockIdx.x * blockDim.x + threadIdx.x;
    if (i >= E * 64) return;
    int e = i >> 6, c = i & 63;
    const int* nd = nodes + (size_t)e * deg;
    float s = 0.0f;
    for (int d = 0; d < deg; ++d) s += x[(size_t)nd[d] * 64 + c];
    agg[i] = s * (1.0f / (float)deg);
}

// ---------------------------------------------------------------------------
// Hedge-message GEMM: z = concat(agg,h) [E,128];  t1 = z@Wf+bf, t2 = z@Wc+bc
// WMMA bf16, f32 accum. Also accumulates column sum / sum-of-squares of t1
// for the training-mode BN that follows.
// ---------------------------------------------------------------------------
__global__ __launch_bounds__(256, 2)
void hedge_gemm_kernel(const float* __restrict__ agg, const float* __restrict__ h,
                       const float* __restrict__ Wf, const float* __restrict__ bf,
                       const float* __restrict__ Wc, const float* __restrict__ bc,
                       float* __restrict__ t1, float* __restrict__ t2,
                       float* __restrict__ stats, int E) {
    __shared__ __align__(32) __bf16 sWf[16 * 32 * 16];
    __shared__ __align__(32) __bf16 sWc[16 * 32 * 16];
    int tid = threadIdx.x;
    stage_weights_packed(Wf, sWf, tid);
    stage_weights_packed(Wc, sWc, tid);
    __syncthreads();

    int wave = tid >> 5, lane = tid & 31;
    int row0 = (blockIdx.x * 8 + wave) * 16;     // 16 rows per wave
    if (row0 >= E) return;                       // wave-uniform exit

    int m  = lane & 15;
    int ar = min(row0 + m, E - 1);
    const float* ra = agg + (size_t)ar * 64;
    const float* rb = h   + (size_t)ar * 64;

    v16bf a[4];
#pragma unroll
    for (int kk = 0; kk < 4; ++kk) a[kk] = load_a_concat(ra, rb, kk * 32, lane);

    v8f zero = {0.f, 0.f, 0.f, 0.f, 0.f, 0.f, 0.f, 0.f};
    v8f cf[4], cc[4];
#pragma unroll
    for (int nt = 0; nt < 4; ++nt) { cf[nt] = zero; cc[nt] = zero; }

#pragma unroll
    for (int nt = 0; nt < 4; ++nt) {
#pragma unroll
        for (int kk = 0; kk < 4; ++kk) {
            v16bf bF = load_b_packed(sWf, kk, nt, lane);
            cf[nt] = __builtin_amdgcn_wmma_f32_16x16x32_bf16(
                false, a[kk], false, bF, (short)0, cf[nt], false, false);
            v16bf bC = load_b_packed(sWc, kk, nt, lane);
            cc[nt] = __builtin_amdgcn_wmma_f32_16x16x32_bf16(
                false, a[kk], false, bC, (short)0, cc[nt], false, false);
        }
    }

    // D layout: lane holds (M = v + 8*(lane>=16), N = nt*16 + lane%16)
    int halfo = (lane & 16) ? 8 : 0;
#pragma unroll
    for (int nt = 0; nt < 4; ++nt) {
        int   c  = nt * 16 + m;
        float bF = bf[c], bC = bc[c];
        float s = 0.0f, s2 = 0.0f;
#pragma unroll
        for (int v = 0; v < 8; ++v) {
            int row = row0 + v + halfo;
            if (row < E) {
                float vf = cf[nt][v] + bF;
                float vc = cc[nt][v] + bC;
                t1[(size_t)row * 64 + c] = vf;
                t2[(size_t)row * 64 + c] = vc;
                s += vf; s2 += vf * vf;
            }
        }
        atomicAdd(&stats[c],      s);
        atomicAdd(&stats[64 + c], s2);
    }
}

// stats[0..63]=sum, [64..127]=sumsq  ->  [128..191]=mu, [192..255]=rsqrt(var+eps)
__global__ void stats_finalize_kernel(float* __restrict__ stats, float inv_rows) {
    int c = threadIdx.x;
    float mu  = stats[c] * inv_rows;
    float var = stats[64 + c] * inv_rows - mu * mu;
    stats[128 + c] = mu;
    stats[192 + c] = rsqrtf(var + EPSV);
}

// hmsg = sigmoid(bn(t1)) * softplus(t2)
__global__ void hmsg_kernel(const float* __restrict__ t1, const float* __restrict__ t2,
                            const float* __restrict__ stats,
                            const float* __restrict__ gf, const float* __restrict__ btf,
                            float* __restrict__ hmsg, int E) {
    int i = blockIdx.x * blockDim.x + threadIdx.x;
    if (i >= E * 64) return;
    int c = i & 63;
    float g = (t1[i] - stats[128 + c]) * stats[192 + c] * gf[c] + btf[c];
    hmsg[i] = sigmoidf_(g) * softplusf_(t2[i]);
}

// ---------------------------------------------------------------------------
// Node-message GEMM over incidences: z2 = concat(x[ni], hmsg[hj]) [nnz,128]
// m = sigmoid(z2@Wf2+bf2)*softplus(z2@Wc2+bc2), scatter-added into o[ni].
// ---------------------------------------------------------------------------
__global__ __launch_bounds__(256, 2)
void node_gemm_kernel(const float* __restrict__ x, const float* __restrict__ hmsg,
                      const int* __restrict__ ni, const int* __restrict__ hj,
                      const float* __restrict__ Wf, const float* __restrict__ bf,
                      const float* __restrict__ Wc, const float* __restrict__ bc,
                      float* __restrict__ o, int nnz) {
    __shared__ __align__(32) __bf16 sWf[16 * 32 * 16];
    __shared__ __align__(32) __bf16 sWc[16 * 32 * 16];
    int tid = threadIdx.x;
    stage_weights_packed(Wf, sWf, tid);
    stage_weights_packed(Wc, sWc, tid);
    __syncthreads();

    int wave = tid >> 5, lane = tid & 31;
    int row0 = (blockIdx.x * 8 + wave) * 16;
    if (row0 >= nnz) return;

    int m  = lane & 15;
    int ar = min(row0 + m, nnz - 1);
    const float* ra = x    + (size_t)ni[ar] * 64;   // gathered node feats
    const float* rb = hmsg + (size_t)hj[ar] * 64;   // gathered hedge message

    v16bf a[4];
#pragma unroll
    for (int kk = 0; kk < 4; ++kk) a[kk] = load_a_concat(ra, rb, kk * 32, lane);

    v8f zero = {0.f, 0.f, 0.f, 0.f, 0.f, 0.f, 0.f, 0.f};
    v8f cf[4], cc[4];
#pragma unroll
    for (int nt = 0; nt < 4; ++nt) { cf[nt] = zero; cc[nt] = zero; }

#pragma unroll
    for (int nt = 0; nt < 4; ++nt) {
#pragma unroll
        for (int kk = 0; kk < 4; ++kk) {
            v16bf bF = load_b_packed(sWf, kk, nt, lane);
            cf[nt] = __builtin_amdgcn_wmma_f32_16x16x32_bf16(
                false, a[kk], false, bF, (short)0, cf[nt], false, false);
            v16bf bC = load_b_packed(sWc, kk, nt, lane);
            cc[nt] = __builtin_amdgcn_wmma_f32_16x16x32_bf16(
                false, a[kk], false, bC, (short)0, cc[nt], false, false);
        }
    }

    int halfo = (lane & 16) ? 8 : 0;
    int atomv[8];
#pragma unroll
    for (int v = 0; v < 8; ++v) {
        int row = row0 + v + halfo;
        atomv[v] = (row < nnz) ? ni[row] : -1;
    }
#pragma unroll
    for (int nt = 0; nt < 4; ++nt) {
        int   c  = nt * 16 + m;
        float bF = bf[c], bC = bc[c];
#pragma unroll
        for (int v = 0; v < 8; ++v) {
            if (atomv[v] >= 0) {
                float mv = sigmoidf_(cf[nt][v] + bF) * softplusf_(cc[nt][v] + bC);
                atomicAdd(&o[(size_t)atomv[v] * 64 + c], mv);
            }
        }
    }
}

// o /= max(cnt,1) and accumulate per-column sum / sumsq for BN.
__global__ void o_finalize_kernel(float* __restrict__ o, const float* __restrict__ cnt,
                                  float* __restrict__ stats, int n) {
    int i = blockIdx.x * blockDim.x + threadIdx.x;
    if (i >= n * 64) return;
    int nd = i >> 6, c = i & 63;
    float v = o[i] / fmaxf(cnt[nd], 1.0f);
    o[i] = v;
    atomicAdd(&stats[c],      v);
    atomicAdd(&stats[64 + c], v * v);
}

// x = relu(x + bn(o))
__global__ void node_update_kernel(float* __restrict__ x, const float* __restrict__ o,
                                   const float* __restrict__ stats,
                                   const float* __restrict__ go, const float* __restrict__ bto,
                                   int n) {
    int i = blockIdx.x * blockDim.x + threadIdx.x;
    if (i >= n * 64) return;
    int c = i & 63;
    float v = x[i] + (o[i] - stats[128 + c]) * stats[192 + c] * go[c] + bto[c];
    x[i] = fmaxf(v, 0.0f);
}

// Crystal pooling: scatter-add node feats by batch id.
__global__ void pool_kernel(const float* __restrict__ x, const int* __restrict__ batch,
                            float* __restrict__ yp) {
    int i = blockIdx.x * blockDim.x + threadIdx.x;
    if (i >= Nn * 64) return;
    int n = i >> 6, c = i & 63;
    atomicAdd(&yp[(size_t)batch[n] * 64 + c], x[i]);
}

__global__ void div_rows_kernel(float* __restrict__ y, const float* __restrict__ cnt,
                                int rows, int cols) {
    int i = blockIdx.x * blockDim.x + threadIdx.x;
    if (i >= rows * cols) return;
    int r = i / cols;
    y[i] /= fmaxf(cnt[r], 1.0f);
}

// ---------------------------------------------------------------------------
static inline int gridFor(long long n, int b) { return (int)((n + b - 1) / b); }

extern "C" void kernel_launch(void* const* d_in, const int* in_sizes, int n_in,
                              void* d_out, int out_size, void* d_ws, size_t ws_size,
                              hipStream_t stream) {
    (void)in_sizes; (void)n_in; (void)out_size; (void)ws_size;

    // setup_inputs() dict order
    const float* atom_x     = (const float*)d_in[0];
    const float* bond_attr  = (const float*)d_in[1];
    const float* motif_attr = (const float*)d_in[2];
    const int*   bond_index  = (const int*)d_in[3];   // [2, 2*EB]
    const int*   motif_index = (const int*)d_in[4];   // [2, 13*M]
    const int*   batch       = (const int*)d_in[5];
    const float* We = (const float*)d_in[6];  const float* be = (const float*)d_in[7];
    const float* Wb = (const float*)d_in[8];  const float* bb = (const float*)d_in[9];
    const float* Wm = (const float*)d_in[10]; const float* bm = (const float*)d_in[11];
    const float* Wf1 = (const float*)d_in[12]; const float* bf1 = (const float*)d_in[13];
    const float* Wc1 = (const float*)d_in[14]; const float* bc1 = (const float*)d_in[15];
    const float* Wf2 = (const float*)d_in[16]; const float* bf2 = (const float*)d_in[17];
    const float* Wc2 = (const float*)d_in[18]; const float* bc2 = (const float*)d_in[19];
    const float* gf  = (const float*)d_in[20]; const float* btf = (const float*)d_in[21];
    const float* go  = (const float*)d_in[22]; const float* bto = (const float*)d_in[23];
    const float* W1 = (const float*)d_in[24]; const float* b1 = (const float*)d_in[25];
    const float* W2 = (const float*)d_in[26]; const float* b2 = (const float*)d_in[27];
    const float* W3 = (const float*)d_in[28]; const float* b3 = (const float*)d_in[29];
    const float* Wo = (const float*)d_in[30]; const float* bo = (const float*)d_in[31];

    const int* bond_nodes   = bond_index;
    const int* bond_hedges  = bond_index + 2 * EBn;
    const int* motif_nodes  = motif_index;
    const int* motif_hedges = motif_index + 13 * Mm;

    // Workspace carve-out (floats)
    float* ws = (float*)d_ws;
    size_t off = 0;
    float* x     = ws + off; off += (size_t)Nn  * 64;
    float* hb    = ws + off; off += (size_t)EBn * 64;
    float* hm    = ws + off; off += (size_t)Mm  * 64;
    float* agg   = ws + off; off += (size_t)EBn * 64;   // reused as hmsg
    float* t1    = ws + off; off += (size_t)EBn * 64;
    float* t2    = ws + off; off += (size_t)EBn * 64;
    float* o     = ws + off; off += (size_t)Nn  * 64;
    float* stats = ws + off; off += 256;
    float* cntNb = ws + off; off += (size_t)Nn;
    float* cntNm = ws + off; off += (size_t)Nn;
    float* cntB  = ws + off; off += (size_t)Bb;
    float* yp    = ws + off; off += (size_t)Bb * 64;
    float* h1    = ws + off; off += (size_t)Bb * 64;
    float* h2    = ws + off; off += (size_t)Bb * 128;
    float* h3    = ws + off; off += (size_t)Bb * 128;

    const int TB = 256;

    // --- launch-invariant segment counts ------------------------------------
    hipMemsetAsync(cntNb, 0, sizeof(float) * Nn, stream);
    hipMemsetAsync(cntNm, 0, sizeof(float) * Nn, stream);
    hipMemsetAsync(cntB,  0, sizeof(float) * Bb, stream);
    count_kernel<<<gridFor(2 * EBn, TB), TB, 0, stream>>>(bond_nodes, 2 * EBn, cntNb);
    count_kernel<<<gridFor(13 * Mm, TB), TB, 0, stream>>>(motif_nodes, 13 * Mm, cntNm);
    count_kernel<<<gridFor(Nn, TB), TB, 0, stream>>>(batch, Nn, cntB);

    // --- embeddings ---------------------------------------------------------
    lin_kernel<<<gridFor((long long)Nn * 64, TB), TB, 0, stream>>>(atom_x, We, be, x, Nn, 92, 64, 0);
    lin_kernel<<<gridFor((long long)EBn * 64, TB), TB, 0, stream>>>(bond_attr, Wb, bb, hb, EBn, 40, 64, 0);
    lin_kernel<<<gridFor((long long)Mm * 64, TB), TB, 0, stream>>>(motif_attr, Wm, bm, hm, Mm, 94, 64, 0);

    // --- 6 CHGConv layers (b, m, b, m, b, m) --------------------------------
    for (int l = 0; l < NLAYERS; ++l) {
        bool isBond = (l % 2 == 0);
        int          E      = isBond ? EBn : Mm;
        int          deg    = isBond ? 2 : 13;
        int          nnz    = E * deg;
        const int*   nodes  = isBond ? bond_nodes : motif_nodes;
        const int*   hedges = isBond ? bond_hedges : motif_hedges;
        const float* h      = isBond ? hb : hm;
        const float* cntN   = isBond ? cntNb : cntNm;

        const float* Wf1l = Wf1 + (size_t)l * 128 * 64; const float* bf1l = bf1 + l * 64;
        const float* Wc1l = Wc1 + (size_t)l * 128 * 64; const float* bc1l = bc1 + l * 64;
        const float* Wf2l = Wf2 + (size_t)l * 128 * 64; const float* bf2l = bf2 + l * 64;
        const float* Wc2l = Wc2 + (size_t)l * 128 * 64; const float* bc2l = bc2 + l * 64;
        const float* gfl  = gf  + l * 64; const float* btfl = btf + l * 64;
        const float* gol  = go  + l * 64; const float* btol = bto + l * 64;

        // hedge message
        hipMemsetAsync(stats, 0, 256 * sizeof(float), stream);
        agg_kernel<<<gridFor((long long)E * 64, TB), TB, 0, stream>>>(x, nodes, agg, E, deg);
        hedge_gemm_kernel<<<gridFor(E, 128), TB, 0, stream>>>(
            agg, h, Wf1l, bf1l, Wc1l, bc1l, t1, t2, stats, E);
        stats_finalize_kernel<<<1, 64, 0, stream>>>(stats, 1.0f / (float)E);
        hmsg_kernel<<<gridFor((long long)E * 64, TB), TB, 0, stream>>>(
            t1, t2, stats, gfl, btfl, agg /* hmsg overwrites agg */, E);

        // node update
        hipMemsetAsync(o, 0, (size_t)Nn * 64 * sizeof(float), stream);
        hipMemsetAsync(stats, 0, 256 * sizeof(float), stream);
        node_gemm_kernel<<<gridFor(nnz, 128), TB, 0, stream>>>(
            x, agg, nodes, hedges, Wf2l, bf2l, Wc2l, bc2l, o, nnz);
        o_finalize_kernel<<<gridFor((long long)Nn * 64, TB), TB, 0, stream>>>(o, cntN, stats, Nn);
        stats_finalize_kernel<<<1, 64, 0, stream>>>(stats, 1.0f / (float)Nn);
        node_update_kernel<<<gridFor((long long)Nn * 64, TB), TB, 0, stream>>>(
            x, o, stats, gol, btol, Nn);
    }

    // --- pooling + MLP head -------------------------------------------------
    hipMemsetAsync(yp, 0, (size_t)Bb * 64 * sizeof(float), stream);
    pool_kernel<<<gridFor((long long)Nn * 64, TB), TB, 0, stream>>>(x, batch, yp);
    div_rows_kernel<<<gridFor((long long)Bb * 64, TB), TB, 0, stream>>>(yp, cntB, Bb, 64);

    lin_kernel<<<gridFor((long long)Bb * 64,  TB), TB, 0, stream>>>(yp, W1, b1, h1, Bb, 64, 64, 1);
    lin_kernel<<<gridFor((long long)Bb * 128, TB), TB, 0, stream>>>(h1, W2, b2, h2, Bb, 64, 128, 1);
    lin_kernel<<<gridFor((long long)Bb * 128, TB), TB, 0, stream>>>(h2, W3, b3, h3, Bb, 128, 128, 1);
    lin_kernel<<<gridFor((long long)Bb * 1,   TB), TB, 0, stream>>>(h3, Wo, bo, (float*)d_out, Bb, 128, 1, 0);
}
